// OneHeadAttention_1151051235982
// MI455X (gfx1250) — compile-verified, matrix operand order checked by reading, not run on hardware
//
#include <hip/hip_runtime.h>
#include <hip/hip_bf16.h>
#include <math.h>

typedef __attribute__((ext_vector_type(16))) _Float16 v16h;
typedef __attribute__((ext_vector_type(8)))  _Float16 v8h;
typedef __attribute__((ext_vector_type(8)))  float    v8f;
typedef __attribute__((ext_vector_type(4)))  float    v4f;
typedef __attribute__((ext_vector_type(4)))  unsigned int v4u;

// ---------------------------------------------------------------------------
// WMMA fragment helpers (wave32, v_wmma_f32_16x16x32_f16).
// 16-bit A layout (ISA 7.12.2): lane L holds row r0+(L&15); VGPRs 0-3 hold
// K = (L<16 ? 0 : 8)..+7, VGPRs 4-7 hold K = (L<16 ? 16 : 24)..+7 — two
// CONTIGUOUS 16-byte runs -> two ds_load_b128 per fragment.
// B operands staged [N][K] row-major in LDS so the same loader serves both.
// C/D f32: n = lane&15, m = r + 8*(lane>=16).
// ---------------------------------------------------------------------------
__device__ __forceinline__ v16h loadfrag_h(const _Float16* base, int r0, int k0,
                                           int stride, unsigned lane) {
  const _Float16* p = base + (r0 + (int)(lane & 15u)) * stride + k0;
  int hi = (lane & 16u) ? 1 : 0;
  v8h lo = *(const v8h*)(p + (hi ? 8 : 0));
  v8h hj = *(const v8h*)(p + (hi ? 24 : 16));
  return __builtin_shufflevector(lo, hj, 0, 1, 2, 3, 4, 5, 6, 7, 8, 9, 10, 11,
                                 12, 13, 14, 15);
}

// A fragment for 2-tap conv chunk: K 0..15 = tap0 ch0..15, K 16..31 = tap1.
__device__ __forceinline__ v16h frag2tap(const _Float16* p0, const _Float16* p1,
                                         unsigned lane) {
  int hi = (lane & 16u) ? 8 : 0;
  v8h lo = *(const v8h*)(p0 + hi);
  v8h hj = *(const v8h*)(p1 + hi);
  return __builtin_shufflevector(lo, hj, 0, 1, 2, 3, 4, 5, 6, 7, 8, 9, 10, 11,
                                 12, 13, 14, 15);
}

// A fragment for 1-tap conv chunk over a 32-channel pixel record.
__device__ __forceinline__ v16h frag1tap(const _Float16* p, unsigned lane) {
  int hi = (lane & 16u) ? 1 : 0;
  v8h lo = *(const v8h*)(p + (hi ? 8 : 0));
  v8h hj = *(const v8h*)(p + (hi ? 24 : 16));
  return __builtin_shufflevector(lo, hj, 0, 1, 2, 3, 4, 5, 6, 7, 8, 9, 10, 11,
                                 12, 13, 14, 15);
}

// A fragment from f32 rows (softmax probabilities): 4x ds_load_b128 + cvt
__device__ __forceinline__ v16h loadfrag_f32(const float* base, int k0,
                                             int stride, unsigned lane) {
  const float* p = base + (int)(lane & 15u) * stride + k0;
  int hi = (lane & 16u) ? 1 : 0;
  v4f c0 = *(const v4f*)(p + (hi ? 8 : 0));
  v4f c1 = *(const v4f*)(p + (hi ? 8 : 0) + 4);
  v4f c2 = *(const v4f*)(p + (hi ? 24 : 16));
  v4f c3 = *(const v4f*)(p + (hi ? 24 : 16) + 4);
  v16h v;
#pragma unroll
  for (int t = 0; t < 4; ++t) {
    v[t]      = (_Float16)c0[t];
    v[4 + t]  = (_Float16)c1[t];
    v[8 + t]  = (_Float16)c2[t];
    v[12 + t] = (_Float16)c3[t];
  }
  return v;
}

__device__ __forceinline__ v8f wmma_f16(v16h a, v16h b, v8f c) {
  return __builtin_amdgcn_wmma_f32_16x16x32_f16(false, a, false, b, (short)0, c,
                                                false, false);
}

__device__ __forceinline__ v8f zero8() {
  v8f z = {0.f, 0.f, 0.f, 0.f, 0.f, 0.f, 0.f, 0.f};
  return z;
}

// ---------------------------------------------------------------------------
// Kernel 1: conv stack as per-tap WMMA GEMMs, one block (8 wave32) per image.
// Activations stored CHANNEL-LAST f16 in LDS so per-tap A-fragments are two
// raw ds_load_b128 from shifted pixel records (no im2col).
// D-stores are UNGUARDED: overflow tiles land in pre-zeroed slack pixels that
// no valid downstream output ever reads (conv3 valid reads <=483 of 512,
// conv4 valid reads <=360 of 368).
// Dynamic LDS layout (bytes):
//   xin  @      0 ( 3136)  28x28 f32
//   ws1  @   3136 ( 1024)  w1 f32
//   bb   @   4160 (  512)  biases b1..b4 padded to 32
//   h1   @   4672 (21504)  [672 pix][16ch] f16 (625 valid, slack zeroed)
//   h2   @  26176 (32768)  [512 pix][32ch] f16 (484 valid, ch pad 20->32)
//   h3   @  58944 (23552)  [368 pix][32ch] f16 (361 valid, ch pad 24->32)
//   Bc2  @  82496 (16384)  8 chunks x [32n][32k] f16 (2 taps/chunk)
//   Bc3  @  98880 (32768)  16 taps x [32n][32k] f16
//   Bc4  @ 131648 (32768)  16 taps x [32n][32k] f16
// total 164416 -> 164608
// ---------------------------------------------------------------------------
__global__ __launch_bounds__(256) void conv_stack_kernel(
    const float* __restrict__ x,
    const float* __restrict__ w1, const float* __restrict__ b1,
    const float* __restrict__ w2, const float* __restrict__ b2,
    const float* __restrict__ w3, const float* __restrict__ b3,
    const float* __restrict__ w4, const float* __restrict__ b4,
    _Float16* __restrict__ feat) {
  extern __shared__ char smem[];
  float*     xin = (float*)smem;
  float*     ws1 = (float*)(smem + 3136);
  float*     bbf = (float*)(smem + 4160);     // [4][32]
  _Float16*  h1  = (_Float16*)(smem + 4672);
  _Float16*  h2  = (_Float16*)(smem + 26176);
  _Float16*  h3  = (_Float16*)(smem + 58944);
  _Float16*  Bc2 = (_Float16*)(smem + 82496);
  _Float16*  Bc3 = (_Float16*)(smem + 98880);
  _Float16*  Bc4 = (_Float16*)(smem + 131648);

  const int tid = threadIdx.x;
  const unsigned lane = tid & 31u;
  const int wv = tid >> 5;
  const int b = blockIdx.x;

  // ---- stage: x, w1, biases, per-tap B matrices; zero h1/h2/h3 ----
  {
    const v4f* sx = (const v4f*)(x + (size_t)b * 784);
    v4f* dx = (v4f*)xin;
    for (int i = tid; i < 196; i += 256) dx[i] = sx[i];
    const v4f* s1 = (const v4f*)w1;  v4f* d1 = (v4f*)ws1;
    for (int i = tid; i < 64; i += 256) d1[i] = s1[i];
    for (int i = tid; i < 128; i += 256) {
      int g = i >> 5, n = i & 31;
      float v = 0.f;
      if (g == 0)      { if (n < 16) v = b1[n]; }
      else if (g == 1) { if (n < 20) v = b2[n]; }
      else if (g == 2) { if (n < 24) v = b3[n]; }
      else             { if (n < 30) v = b4[n]; }
      bbf[i] = v;
    }
    // zero h1..h3 (contiguous 77824 B = 4864 x b128)
    v4u zz = {0u, 0u, 0u, 0u};
    v4u* z = (v4u*)h1;
    for (int i = tid; i < 4864; i += 256) z[i] = zz;
    // Bc2: chunk c pairs taps 2c,2c+1; k<16 -> tap 2c ic=k, k>=16 -> tap 2c+1
    for (int idx = tid; idx < 8192; idx += 256) {
      int c = idx >> 10, rem = idx & 1023, n = rem >> 5, k = rem & 31;
      _Float16 v = (_Float16)0.f;
      if (n < 20) {
        int tap = 2 * c + (k >> 4), ic = k & 15;
        v = (_Float16)w2[(n * 16 + ic) * 16 + tap];
      }
      Bc2[idx] = v;
    }
    // Bc3: tap t, k = ic (20 valid of 32), n = oc (24 valid of 32)
    for (int idx = tid; idx < 16384; idx += 256) {
      int t = idx >> 10, rem = idx & 1023, n = rem >> 5, k = rem & 31;
      Bc3[idx] = (n < 24 && k < 20) ? (_Float16)w3[(n * 20 + k) * 16 + t]
                                    : (_Float16)0.f;
    }
    // Bc4: tap t, k = ic (24 valid of 32), n = oc (30 valid of 32)
    for (int idx = tid; idx < 16384; idx += 256) {
      int t = idx >> 10, rem = idx & 1023, n = rem >> 5, k = rem & 31;
      Bc4[idx] = (n < 30 && k < 24) ? (_Float16)w4[(n * 24 + k) * 16 + t]
                                    : (_Float16)0.f;
    }
  }
  __syncthreads();

  // ---- conv1 (ic=1, scalar): 28x28 -> 16x25x25, channel-last f16 ----
  for (int idx = tid; idx < 10000; idx += 256) {
    int p = idx >> 4, oc = idx & 15;
    int oy = p / 25, ox = p - oy * 25;
    float acc = bbf[oc];
    const float* wp = ws1 + oc * 16;
#pragma unroll
    for (int ky = 0; ky < 4; ++ky)
#pragma unroll
      for (int kx = 0; kx < 4; ++kx)
        acc += xin[(oy + ky) * 28 + ox + kx] * wp[ky * 4 + kx];
    h1[idx] = (_Float16)fmaxf(acc, 0.f);  // idx == p*16 + oc
  }
  __syncthreads();

  // ---- conv2 via WMMA: M=484(->496) x N=20(->32) x K=256 (8 chunks) ----
  for (int u = wv; u < 62; u += 8) {
    int mt = u >> 1, nt = u & 1;
    int m = mt * 16 + (int)(lane & 15u);
    int oy = m / 22, ox = m - oy * 22;
    v8f acc = zero8();
#pragma unroll
    for (int c = 0; c < 8; ++c) {
      int t0 = 2 * c, t1 = 2 * c + 1;
      const _Float16* p0 = h1 + ((oy + (t0 >> 2)) * 25 + ox + (t0 & 3)) * 16;
      const _Float16* p1 = h1 + ((oy + (t1 >> 2)) * 25 + ox + (t1 & 3)) * 16;
      v16h a = frag2tap(p0, p1, lane);
      v16h bf = loadfrag_h(Bc2 + c * 1024, nt * 16, 0, 32, lane);
      acc = wmma_f16(a, bf, acc);
    }
    int n = nt * 16 + (int)(lane & 15u);
    float bias = bbf[32 + n];
    int mh = (lane & 16u) ? 8 : 0;
#pragma unroll
    for (int r = 0; r < 8; ++r)
      h2[(mt * 16 + mh + r) * 32 + n] = (_Float16)fmaxf(acc[r] + bias, 0.f);
  }
  __syncthreads();

  // ---- conv3 via WMMA: M=361(->368) x N=24(->32) x K=512 (16 taps) ----
  for (int u = wv; u < 46; u += 8) {
    int mt = u >> 1, nt = u & 1;
    int m = mt * 16 + (int)(lane & 15u);
    int oy = m / 19, ox = m - oy * 19;
    v8f acc = zero8();
#pragma unroll
    for (int t = 0; t < 16; ++t) {
      const _Float16* p = h2 + ((oy + (t >> 2)) * 22 + ox + (t & 3)) * 32;
      v16h a = frag1tap(p, lane);
      v16h bf = loadfrag_h(Bc3 + t * 1024, nt * 16, 0, 32, lane);
      acc = wmma_f16(a, bf, acc);
    }
    int n = nt * 16 + (int)(lane & 15u);
    float bias = bbf[64 + n];
    int mh = (lane & 16u) ? 8 : 0;
#pragma unroll
    for (int r = 0; r < 8; ++r)
      h3[(mt * 16 + mh + r) * 32 + n] = (_Float16)fmaxf(acc[r] + bias, 0.f);
  }
  __syncthreads();

  // ---- conv4 via WMMA: M=256 x N=30(->32) x K=512; write features+coords ----
  _Float16* fb = feat + (size_t)b * 8192;
  for (int u = wv; u < 32; u += 8) {
    int mt = u >> 1, nt = u & 1;
    int m = mt * 16 + (int)(lane & 15u);
    int oy = m >> 4, ox = m & 15;
    v8f acc = zero8();
#pragma unroll
    for (int t = 0; t < 16; ++t) {
      const _Float16* p = h3 + ((oy + (t >> 2)) * 19 + ox + (t & 3)) * 32;
      v16h a = frag1tap(p, lane);
      v16h bf = loadfrag_h(Bc4 + t * 1024, nt * 16, 0, 32, lane);
      acc = wmma_f16(a, bf, acc);
    }
    int n = nt * 16 + (int)(lane & 15u);
    int mh = (lane & 16u) ? 8 : 0;
    if (n < 30) {  // per-lane-uniform guard hoisted outside the r-loop
      float bias = bbf[96 + n];
#pragma unroll
      for (int r = 0; r < 8; ++r)
        fb[(mt * 16 + mh + r) * 32 + n] = (_Float16)fmaxf(acc[r] + bias, 0.f);
    }
  }
  {
    int n = tid;  // coord channels (columns 30/31, disjoint from tile stores)
    int xx = n % 16, yy = n / 16;
    fb[n * 32 + 30] = (_Float16)((float)xx * 0.0625f);
    fb[n * 32 + 31] = (_Float16)((float)yy * 0.0625f);
  }
}

// ---------------------------------------------------------------------------
// Kernel 2: attention head, one block (8 wave32) per batch element.
// b128 fragment loads, transposed V, softmax normalization folded into the
// accumulator scaling after adj@V.
// ---------------------------------------------------------------------------
__global__ __launch_bounds__(256) void attn_kernel(
    const _Float16* __restrict__ feat,
    const float* __restrict__ kW, const float* __restrict__ kb,
    const float* __restrict__ qW, const float* __restrict__ qb,
    const float* __restrict__ vW, const float* __restrict__ vb,
    const float* __restrict__ kg, const float* __restrict__ kbe,
    const float* __restrict__ qg, const float* __restrict__ qbe,
    const float* __restrict__ vg, const float* __restrict__ vbe,
    const float* __restrict__ l1W, const float* __restrict__ l1b,
    const float* __restrict__ l2W, const float* __restrict__ l2b,
    float* __restrict__ out) {
  extern __shared__ char smem[];
  _Float16* hbuf = (_Float16*)smem;
  _Float16* Qb   = (_Float16*)(smem + 16384);
  _Float16* Kb   = (_Float16*)(smem + 49152);
  _Float16* Vt   = (_Float16*)(smem + 81920);
  _Float16* wbuf = (_Float16*)(smem + 107264);
  float* tmpf = (float*)(smem + 111104);
  float* sc   = (float*)(smem + 160256);
  float* red  = (float*)(smem + 293376);
  float* mxv  = (float*)(smem + 295456);
  float* ov   = (float*)(smem + 295648);
  float* bb   = (float*)(smem + 295712);

  const int tid = threadIdx.x;
  const unsigned lane = tid & 31u;
  const int wv = tid >> 5;
  const int b = blockIdx.x;

  {
    const v4u* src = (const v4u*)(feat + (size_t)b * 8192);
    v4u* dst = (v4u*)hbuf;
    for (int i = tid; i < 1024; i += 256) dst[i] = src[i];
    const float* Bp[3] = {vb, kb, qb};
    for (int idx = tid; idx < 144; idx += 256) {
      int p = idx / 48, j = idx % 48;
      bb[idx] = (j < 36) ? Bp[p][j] : 0.f;
    }
  }
  __syncthreads();

  const float* Ws[3] = {vW, kW, qW};
  const float* Gs[3] = {vg, kg, qg};
  const float* Es[3] = {vbe, kbe, qbe};

  for (int p = 0; p < 3; ++p) {
    for (int idx = tid; idx < 48 * 40; idx += 256) {
      int n = idx / 40, k = idx % 40;
      wbuf[idx] = (n < 36 && k < 32) ? (_Float16)Ws[p][n * 32 + k]
                                     : (_Float16)0.f;
    }
    __syncthreads();

    for (int t = wv; t < 48; t += 8) {
      int rt = t / 3, nt = t % 3;
      v16h a  = loadfrag_h(hbuf, rt * 16, 0, 32, lane);
      v16h bf = loadfrag_h(wbuf, nt * 16, 0, 40, lane);
      v8f acc = zero8();
      acc = wmma_f16(a, bf, acc);
      int n = nt * 16 + (int)(lane & 15u);
      float bias = bb[p * 48 + n];
      int mh = (lane & 16u) ? 8 : 0;
#pragma unroll
      for (int r = 0; r < 8; ++r)
        tmpf[(rt * 16 + mh + r) * 48 + n] = acc[r] + bias;
    }
    __syncthreads();

    {
      float s = 0.f, s2 = 0.f;
      const float* rowp = tmpf + tid * 48;
#pragma unroll
      for (int c = 0; c < 36; c += 4) {
        v4f v = *(const v4f*)(rowp + c);
        s += v[0] + v[1] + v[2] + v[3];
        s2 += v[0] * v[0] + v[1] * v[1] + v[2] * v[2] + v[3] * v[3];
      }
      red[tid] = s; red[256 + tid] = s2;
    }
    __syncthreads();
    for (int off = 128; off > 0; off >>= 1) {
      if (tid < off) {
        red[tid] += red[tid + off];
        red[256 + tid] += red[256 + tid + off];
      }
      __syncthreads();
    }
    float mu   = red[0] * (1.f / 9216.f);
    float var  = red[256] * (1.f / 9216.f) - mu * mu;
    float rstd = rsqrtf(var + 1e-5f);

    if (p == 0) {
      for (int idx = tid; idx < 48 * 256; idx += 256) {
        int j = idx >> 8, node = idx & 255;
        float o = 0.f;
        if (j < 36) {
          float v = (tmpf[node * 48 + j] - mu) * rstd;
          o = v * Gs[0][node * 36 + j] + Es[0][node * 36 + j];
        }
        Vt[j * 264 + node] = (_Float16)o;
      }
    } else {
      _Float16* dst = (p == 1) ? Kb : Qb;
      for (int idx = tid; idx < 256 * 64; idx += 256) {
        int n = idx >> 6, j = idx & 63;
        float o = 0.f;
        if (j < 36) {
          float v = (tmpf[n * 48 + j] - mu) * rstd;
          o = v * Gs[p][n * 36 + j] + Es[p][n * 36 + j];
        }
        dst[idx] = (_Float16)o;
      }
    }
    __syncthreads();
  }

  const float scale = 0.16666667f;  // 1/sqrt(36)
  float* scw = sc + wv * 16 * 260;
  for (int blk = wv; blk < 16; blk += 8) {
    int row0 = blk * 16;
    v16h qa0 = loadfrag_h(Qb, row0, 0, 64, lane);
    v16h qa1 = loadfrag_h(Qb, row0, 32, 64, lane);
    for (int ct = 0; ct < 16; ++ct) {
      v8f acc = zero8();
      acc = wmma_f16(qa0, loadfrag_h(Kb, ct * 16, 0, 64, lane), acc);
      acc = wmma_f16(qa1, loadfrag_h(Kb, ct * 16, 32, 64, lane), acc);
      int n = ct * 16 + (int)(lane & 15u);
      int mh = (lane & 16u) ? 8 : 0;
#pragma unroll
      for (int r = 0; r < 8; ++r)
        scw[(mh + r) * 260 + n] = acc[r] * scale;
    }
    if (lane < 16u) {
      float* rowp = scw + (int)lane * 260;
      float mx = -1e30f;
      for (int c = 0; c < 256; c += 4) {
        v4f v = *(const v4f*)(rowp + c);
        mx = fmaxf(mx, fmaxf(fmaxf(v[0], v[1]), fmaxf(v[2], v[3])));
      }
      float s = 0.f;
      for (int c = 0; c < 256; c += 4) {
        v4f v = *(const v4f*)(rowp + c);
        v4f e;
        e[0] = __expf(v[0] - mx); e[1] = __expf(v[1] - mx);
        e[2] = __expf(v[2] - mx); e[3] = __expf(v[3] - mx);
        s += e[0] + e[1] + e[2] + e[3];
        *(v4f*)(rowp + c) = e;
      }
      rowp[256] = s;
    }
    v8f av0 = zero8(), av1 = zero8(), av2 = zero8();
    for (int kc = 0; kc < 8; ++kc) {
      v16h a = loadfrag_f32(scw, kc * 32, 260, lane);
      av0 = wmma_f16(a, loadfrag_h(Vt, 0, kc * 32, 264, lane), av0);
      av1 = wmma_f16(a, loadfrag_h(Vt, 16, kc * 32, 264, lane), av1);
      av2 = wmma_f16(a, loadfrag_h(Vt, 32, kc * 32, 264, lane), av2);
    }
    int mh = (lane & 16u) ? 8 : 0;
    int nl = (int)(lane & 15u);
#pragma unroll
    for (int r = 0; r < 8; ++r) {
      float invr = 1.f / scw[(mh + r) * 260 + 256];
      tmpf[(row0 + mh + r) * 48 + nl]      = av0[r] * invr;
      tmpf[(row0 + mh + r) * 48 + 16 + nl] = av1[r] * invr;
      tmpf[(row0 + mh + r) * 48 + 32 + nl] = av2[r] * invr;
    }
  }
  __syncthreads();

  float* l1Ws = (float*)Qb;  // Qb region is dead now
  for (int i = tid; i < 1296; i += 256) l1Ws[i] = l1W[i];
  __syncthreads();

  float* att2 = sc;
  {
    const float* ar = tmpf + tid * 48;
    for (int j = 0; j < 36; ++j) {
      float s = l1b[j];
      const float* wr = l1Ws + j * 36;
      for (int k = 0; k < 36; ++k) s += ar[k] * wr[k];
      att2[tid * 40 + j] = fmaxf(s, 0.f);
    }
  }
  __syncthreads();

  {
    float s = 0.f, s2 = 0.f;
    const float* rp = att2 + tid * 40;
#pragma unroll
    for (int c = 0; c < 36; c += 4) {
      v4f v = *(const v4f*)(rp + c);
      s += v[0] + v[1] + v[2] + v[3];
      s2 += v[0] * v[0] + v[1] * v[1] + v[2] * v[2] + v[3] * v[3];
    }
    red[tid] = s; red[256 + tid] = s2;
  }
  __syncthreads();
  for (int off = 128; off > 0; off >>= 1) {
    if (tid < off) {
      red[tid] += red[tid + off];
      red[256 + tid] += red[256 + tid + off];
    }
    __syncthreads();
  }
  float mu   = red[0] * (1.f / 9216.f);
  float var  = red[256] * (1.f / 9216.f) - mu * mu;
  float rstd = rsqrtf(var + 1e-5f);

  if (tid < 36) {
    float m = -1e30f;
    for (int n = 0; n < 256; ++n) m = fmaxf(m, att2[n * 40 + tid]);
    mxv[tid] = (m - mu) * rstd;
  }
  __syncthreads();

  if (tid < 10) {
    float s = l2b[tid];
    for (int j = 0; j < 36; ++j) s += mxv[j] * l2W[tid * 36 + j];
    ov[tid] = s;
  }
  __syncthreads();
  if (tid == 0) {
    float mx = -1e30f;
    for (int o = 0; o < 10; ++o) mx = fmaxf(mx, ov[o]);
    float s = 0.f;
    for (int o = 0; o < 10; ++o) s += __expf(ov[o] - mx);
    red[0] = mx + __logf(s);
  }
  __syncthreads();
  if (tid < 10) out[(size_t)b * 10 + tid] = ov[tid] - red[0];
}

// ---------------------------------------------------------------------------
extern "C" void kernel_launch(void* const* d_in, const int* in_sizes, int n_in,
                              void* d_out, int out_size, void* d_ws, size_t ws_size,
                              hipStream_t stream) {
  const float* x  = (const float*)d_in[0];
  const float* w1 = (const float*)d_in[1];  const float* b1 = (const float*)d_in[2];
  const float* w2 = (const float*)d_in[3];  const float* b2 = (const float*)d_in[4];
  const float* w3 = (const float*)d_in[5];  const float* b3 = (const float*)d_in[6];
  const float* w4 = (const float*)d_in[7];  const float* b4 = (const float*)d_in[8];
  const float* kW = (const float*)d_in[9];  const float* kbv = (const float*)d_in[10];
  const float* qW = (const float*)d_in[11]; const float* qbv = (const float*)d_in[12];
  const float* vW = (const float*)d_in[13]; const float* vbv = (const float*)d_in[14];
  const float* kg = (const float*)d_in[15]; const float* kbe = (const float*)d_in[16];
  const float* qg = (const float*)d_in[17]; const float* qbe = (const float*)d_in[18];
  const float* vg = (const float*)d_in[19]; const float* vbe = (const float*)d_in[20];
  const float* l1W = (const float*)d_in[21]; const float* l1b = (const float*)d_in[22];
  const float* l2W = (const float*)d_in[23]; const float* l2b = (const float*)d_in[24];

  const int B = in_sizes[0] / (28 * 28);   // 2048
  _Float16* feat = (_Float16*)d_ws;        // [B][256][32] f16 = 16 KB/batch

  conv_stack_kernel<<<B, 256, 164608, stream>>>(x, w1, b1, w2, b2, w3, b3, w4,
                                                b4, feat);
  attn_kernel<<<B, 256, 296448, stream>>>(feat, kW, kbv, qW, qbv, vW, vbv,
                                          kg, kbe, qg, qbe, vg, vbe,
                                          l1W, l1b, l2W, l2b, (float*)d_out);
}
